// TopKPooler_81965155877090
// MI455X (gfx1250) — compile-verified, matrix operand order checked by reading, not run on hardware
//
#include <hip/hip_runtime.h>
#include <math.h>

// Problem constants (match reference)
#define NN 200000
#define EE 600000
#define GG 64
#define DD 128
#define KCH 4          // 128 / 32 K-chunks for wmma f16 16x16x32
#define NTL 8          // 128 / 16 column tiles
#define MAXG 4096      // per-graph node capacity (binomial(200000,1/64) max ~3400)

typedef __attribute__((ext_vector_type(16))) _Float16 v16h;
typedef __attribute__((ext_vector_type(8)))  float    v8f;
typedef __attribute__((ext_vector_type(4)))  int      v4i;

typedef __attribute__((address_space(1))) v4i as1_v4i;   // global int4
typedef __attribute__((address_space(3))) v4i as3_v4i;   // LDS int4

#if __has_builtin(__builtin_amdgcn_global_load_async_to_lds_b128)
#define HAVE_ASYNC_LDS 1
#else
#define HAVE_ASYNC_LDS 0
#endif

// ---------------------------------------------------------------- prep kernels

// inv ||pool_w||, zero per-graph counters
__global__ void k_prep(const float* __restrict__ pool_w, float* __restrict__ invn,
                       int* __restrict__ cnt) {
  __shared__ float red[128];
  int t = threadIdx.x;                       // blockDim = 128
  float w = pool_w[t];
  red[t] = w * w;
  __syncthreads();
  for (int s = 64; s > 0; s >>= 1) { if (t < s) red[t] += red[t + s]; __syncthreads(); }
  if (t == 0) invn[0] = rsqrtf(red[0]);
  if (t < GG) cnt[t] = 0;
}

__global__ void k_count(const int* __restrict__ batch, int* __restrict__ cnt) {
  int i = blockIdx.x * blockDim.x + threadIdx.x;
  if (i < NN) atomicAdd(&cnt[batch[i]], 1);
}

__global__ void k_scan(const int* __restrict__ cnt, int* __restrict__ start) {
  if (threadIdx.x == 0) {
    int a = 0;
    for (int g = 0; g < GG; ++g) { start[g] = a; a += cnt[g]; }
  }
}

__global__ void k_initvalid(int* __restrict__ valid) {
  int i = blockIdx.x * blockDim.x + threadIdx.x;
  if (i < NN) valid[i] = 1;
}

// Pack gcn_w (f32 [K=128][N=128] row-major) into f16 B-operand tiles laid out
// exactly per-lane for v_wmma_f32_16x16x32_f16:
//   lane l holds column n = (l&15); halfs 0..7  -> K = kbase + (l>>4)*8 + h
//                                   halfs 8..15 -> K = kbase + 16 + (l>>4)*8 + (h-8)
__global__ void k_packB(const float* __restrict__ W, _Float16* __restrict__ pk) {
  int id = blockIdx.x * blockDim.x + threadIdx.x;   // KCH*NTL*32 = 1024 items
  if (id >= KCH * NTL * 32) return;
  int l = id & 31;
  int t = (id >> 5) & 7;
  int c = id >> 8;
  int n  = (t << 4) + (l & 15);
  int kb = (c << 5) + ((l >> 4) << 3);
  _Float16* o = pk + (size_t)id * 16;
  #pragma unroll
  for (int h = 0; h < 16; ++h) {
    int k = (h < 8) ? (kb + h) : (kb + 16 + (h - 8));
    o[h] = (_Float16)W[k * DD + n];
  }
}

// ------------------------------------------------------------- per-iteration

// score s[i] = tanh(x_i . w / ||w||); one wave per node, float4 per lane
__global__ void k_score(const float* __restrict__ x, const float* __restrict__ w,
                        const float* __restrict__ invn, float* __restrict__ s) {
  int wid  = (blockIdx.x * blockDim.x + threadIdx.x) >> 5;
  int lane = threadIdx.x & 31;
  if (wid >= NN) return;
  const float4 a = ((const float4*)(x + (size_t)wid * DD))[lane];
  const float4 b = ((const float4*)w)[lane];
  float d = a.x * b.x + a.y * b.y + a.z * b.z + a.w * b.w;
  for (int o = 16; o > 0; o >>= 1) d += __shfl_xor(d, o, 32);
  if (lane == 0) s[wid] = tanhf(d * invn[0]);
}

// TopK select: one block per graph (nodes contiguous since batch is sorted).
// Exact rank counting with reference tie-break (score desc, index asc).
// Writes valid mask and gate (sel ? s : 0) back into s.
__global__ __launch_bounds__(1024)
void k_select(float* __restrict__ s, int* __restrict__ valid,
              const int* __restrict__ start, const int* __restrict__ cnt) {
  __shared__ float ls[MAXG];
  __shared__ int nvS;
  const float NEG = -3.0e38f;
  int g  = blockIdx.x;
  int st = start[g];
  int c  = cnt[g]; if (c > MAXG) c = MAXG;
  if (threadIdx.x == 0) nvS = 0;
  __syncthreads();
  int myc = 0;
  for (int j = threadIdx.x; j < c; j += blockDim.x) {
    int i = st + j;
    int v = valid[i];
    float sv = v ? s[i] : NEG;
    ls[j] = sv;
    myc += v;
  }
  if (myc) atomicAdd(&nvS, myc);
  __syncthreads();
  int k = (4 * nvS + 4) / 5;                 // ceil(0.8 * n_valid)
  int   nm = 0;
  int   myJ[4]; float myS[4]; int myR[4];
  for (int j = threadIdx.x; j < c; j += blockDim.x) {
    myJ[nm] = j; myS[nm] = ls[j]; myR[nm] = 0; ++nm;
  }
  for (int j2 = 0; j2 < c; ++j2) {
    float o = ls[j2];
    #pragma unroll
    for (int t = 0; t < 4; ++t) {
      if (t < nm) myR[t] += (o > myS[t]) || (o == myS[t] && j2 < myJ[t]);
    }
  }
  for (int t = 0; t < nm; ++t) {
    int i = st + myJ[t];
    if (myS[t] == NEG) { valid[i] = 0; s[i] = 0.f; }
    else {
      int sel = myR[t] < k;
      valid[i] = sel;
      s[i] = sel ? myS[t] : 0.f;            // gate folded for the GEMM
    }
  }
}

// xw = (gate .* x) @ W via v_wmma_f32_16x16x32_f16.
// Packed B (32 KB) is staged into LDS once per block via the gfx1250 async
// global->LDS path (ASYNCcnt) when available; WMMA B operands then feed from
// ds_load_b128 while the global pipe streams A rows.
__global__ __launch_bounds__(256)
void k_gemm(const float* __restrict__ x, const float* __restrict__ gate,
            const _Float16* __restrict__ pk, float* __restrict__ y) {
  __shared__ _Float16 ldsB[KCH * NTL * 32 * 16];   // 32 KB
  {
    int tid = threadIdx.x;
#if HAVE_ASYNC_LDS
    #pragma unroll
    for (int q = 0; q < 8; ++q) {
      int byte = (tid + q * 256) * 16;
      __builtin_amdgcn_global_load_async_to_lds_b128(
          (as1_v4i*)((const char*)pk + byte),
          (as3_v4i*)((char*)ldsB + byte), 0, 0);
    }
#if __has_builtin(__builtin_amdgcn_s_wait_asynccnt)
    __builtin_amdgcn_s_wait_asynccnt(0);
#else
    asm volatile("s_wait_asynccnt 0x0" ::: "memory");
#endif
#else
    const float4* gsrc = (const float4*)pk;
    float4*       ldst = (float4*)ldsB;
    #pragma unroll
    for (int q = 0; q < 8; ++q) ldst[tid + q * 256] = gsrc[tid + q * 256];
#endif
    __syncthreads();
  }

  int wid  = (blockIdx.x * blockDim.x + threadIdx.x) >> 5;
  int lane = threadIdx.x & 31;
  if (wid < NN / 16) {
    int rowbase = wid << 4;
    int m   = lane & 15;
    int row = rowbase + m;
    float gsc = gate[row];
    const float* xr = x + (size_t)row * DD;
    int hiofs = (lane >> 4) << 3;             // 0 or 8
    v8f acc[NTL];
    #pragma unroll
    for (int t = 0; t < NTL; ++t) acc[t] = (v8f){0.f,0.f,0.f,0.f,0.f,0.f,0.f,0.f};
    #pragma unroll
    for (int c = 0; c < KCH; ++c) {
      int kb = (c << 5) + hiofs;
      v16h a;
      #pragma unroll
      for (int h = 0; h < 8; ++h) a[h]     = (_Float16)(gsc * xr[kb + h]);
      #pragma unroll
      for (int h = 0; h < 8; ++h) a[8 + h] = (_Float16)(gsc * xr[kb + 16 + h]);
      const _Float16* bbase = ldsB + ((size_t)c * NTL * 32 + lane) * 16;
      #pragma unroll
      for (int t = 0; t < NTL; ++t) {
        v16h b = *(const v16h*)(bbase + (size_t)t * 32 * 16);
        acc[t] = __builtin_amdgcn_wmma_f32_16x16x32_f16(
            false, a, false, b, (short)0, acc[t], false, false);
      }
    }
    // C/D layout: lane holds col n=(lane&15); vgpr v holds row m = v + 8*(lane>>4)
    #pragma unroll
    for (int t = 0; t < NTL; ++t) {
      float* yb = y + (size_t)rowbase * DD + t * 16 + (lane & 15);
      #pragma unroll
      for (int v = 0; v < 8; ++v) yb[(size_t)(v + hiofs) * DD] = acc[t][v];
    }
  }
}

// degree (with self loop) and dinv
__global__ void k_deginit(const int* __restrict__ valid, float* __restrict__ deg) {
  int i = blockIdx.x * blockDim.x + threadIdx.x;
  if (i < NN) deg[i] = valid[i] ? 1.0f : 0.0f;
}
__global__ void k_degedge(const int* __restrict__ ei, const int* __restrict__ valid,
                          float* __restrict__ deg) {
  int e = blockIdx.x * blockDim.x + threadIdx.x;
  if (e >= EE) return;
  int s = ei[e], d = ei[EE + e];
  if (valid[s] && valid[d]) atomicAdd(&deg[d], 1.0f);
}
__global__ void k_dinv(float* __restrict__ deg) {
  int i = blockIdx.x * blockDim.x + threadIdx.x;
  if (i < NN) { float dg = deg[i]; deg[i] = dg > 0.f ? rsqrtf(dg) : 0.f; }
}

// self-loop init: agg (in xbuf) = dinv^2 * xw
__global__ void k_selfinit(const float* __restrict__ xw, const float* __restrict__ dinv,
                           float* __restrict__ xb) {
  int idx = blockIdx.x * blockDim.x + threadIdx.x;
  if (idx >= NN * DD) return;
  float di = dinv[idx >> 7];
  xb[idx] = di * di * xw[idx];
}

// edge aggregation: one wave per edge, f32 atomics (coef self-masks invalid edges)
__global__ void k_edgeagg(const int* __restrict__ ei, const float* __restrict__ dinv,
                          const float* __restrict__ xw, float* __restrict__ xb) {
  int wid  = (blockIdx.x * blockDim.x + threadIdx.x) >> 5;
  int lane = threadIdx.x & 31;
  if (wid >= EE) return;
  int s = ei[wid], d = ei[EE + wid];
  float coef = dinv[s] * dinv[d];
  if (coef != 0.f) {
    const float* xs = xw + (size_t)s * DD;
    float*       xd = xb + (size_t)d * DD;
    #pragma unroll
    for (int t = 0; t < 4; ++t) {
      int j = lane + 32 * t;
      atomicAdd(&xd[j], xs[j] * coef);
    }
  }
}

// x = valid ? relu(agg + b) : 0
__global__ void k_epilogue(float* __restrict__ xb, const float* __restrict__ bias,
                           const int* __restrict__ valid) {
  int idx = blockIdx.x * blockDim.x + threadIdx.x;
  if (idx >= NN * DD) return;
  float v = xb[idx] + bias[idx & 127];
  xb[idx] = valid[idx >> 7] ? fmaxf(v, 0.f) : 0.f;
}

// per-graph mean pool (block per graph, column-parallel, no atomics)
__global__ void k_pool(const float* __restrict__ xb, const int* __restrict__ valid,
                       const int* __restrict__ start, const int* __restrict__ cnt,
                       float* __restrict__ out) {
  int g = blockIdx.x, j = threadIdx.x;       // blockDim = 128
  int st = start[g], c = cnt[g];
  float sum = 0.f; int vc = 0;
  for (int r = 0; r < c; ++r) {
    int i = st + r;
    vc  += valid[i];                         // broadcast load
    sum += xb[(size_t)i * DD + j];           // invalid rows are zero
  }
  out[g * DD + j] = sum / (float)(vc > 0 ? vc : 1);
}

// -------------------------------------------------------------------- launch

extern "C" void kernel_launch(void* const* d_in, const int* in_sizes, int n_in,
                              void* d_out, int out_size, void* d_ws, size_t ws_size,
                              hipStream_t stream) {
  const float* node_states = (const float*)d_in[0];
  const int*   edge_index  = (const int*)d_in[1];
  const int*   batch       = (const int*)d_in[2];
  const float* pool_w      = (const float*)d_in[3];
  const float* gcn_w       = (const float*)d_in[4];
  const float* gcn_b       = (const float*)d_in[5];
  float* out = (float*)d_out;
  (void)in_sizes; (void)n_in; (void)out_size; (void)ws_size;

  char* w = (char*)d_ws;
  float* xbuf = (float*)w;  w += (size_t)NN * DD * 4;   // x / agg (ping)
  float* xw   = (float*)w;  w += (size_t)NN * DD * 4;   // x @ W
  float* sbuf = (float*)w;  w += (size_t)NN * 4;        // score -> gate
  float* degv = (float*)w;  w += (size_t)NN * 4;        // deg -> dinv
  int*   valid= (int*)w;    w += (size_t)NN * 4;
  int*   cnt  = (int*)w;    w += GG * 4;
  int*   start= (int*)w;    w += GG * 4;
  float* invn = (float*)w;  w += 64;                    // padded scalar
  _Float16* pk = (_Float16*)w;                          // 32 KB packed B

  // one-time prep (re-run each call for determinism)
  (void)hipMemcpyAsync(xbuf, node_states, (size_t)NN * DD * 4,
                       hipMemcpyDeviceToDevice, stream);
  k_prep<<<1, 128, 0, stream>>>(pool_w, invn, cnt);
  k_count<<<(NN + 255) / 256, 256, 0, stream>>>(batch, cnt);
  k_scan<<<1, 64, 0, stream>>>(cnt, start);
  k_initvalid<<<(NN + 255) / 256, 256, 0, stream>>>(valid);
  k_packB<<<4, 256, 0, stream>>>(gcn_w, pk);

  const int ND_BLKS = (NN * DD + 255) / 256;
  for (int it = 0; it < 3; ++it) {
    k_score   <<<(NN * 32 + 255) / 256, 256, 0, stream>>>(xbuf, pool_w, invn, sbuf);
    k_select  <<<GG, 1024, 0, stream>>>(sbuf, valid, start, cnt);
    k_gemm    <<<((NN / 16) * 32 + 255) / 256, 256, 0, stream>>>(xbuf, sbuf, pk, xw);
    k_deginit <<<(NN + 255) / 256, 256, 0, stream>>>(valid, degv);
    k_degedge <<<(EE + 255) / 256, 256, 0, stream>>>(edge_index, valid, degv);
    k_dinv    <<<(NN + 255) / 256, 256, 0, stream>>>(degv);
    k_selfinit<<<ND_BLKS, 256, 0, stream>>>(xw, degv, xbuf);
    k_edgeagg <<<(EE * 32 + 255) / 256, 256, 0, stream>>>(edge_index, degv, xw, xbuf);
    k_epilogue<<<ND_BLKS, 256, 0, stream>>>(xbuf, gcn_b, valid);
  }
  k_pool<<<GG, 128, 0, stream>>>(xbuf, valid, start, cnt, out);
}
